// Transition_88012469829758
// MI455X (gfx1250) — matrix operand order repaired and checked by reading, not compiled, not run on hardware
//
#include <hip/hip_runtime.h>
#include <hip/hip_bf16.h>

#define Bn   8
#define Sn   4096
#define Dn   1024
#define DIMn 512
#define LSTR 1025                 // 32-row logits strip stride (dwords), %64==1 -> conflict-free columns
#define INV_SQRT_D 0.044194173824159216f

typedef __attribute__((ext_vector_type(16))) __bf16 bf16x16;
typedef __attribute__((ext_vector_type(8)))  __bf16 bf16x8;
typedef __attribute__((ext_vector_type(8)))  float  f32x8;

// ---- CDNA5 async global->LDS (guarded; fallback = synchronous bf16 copy) --
#if defined(__has_builtin)
#if __has_builtin(__builtin_amdgcn_global_load_async_to_lds_b128) && \
    __has_builtin(__builtin_amdgcn_s_wait_asynccnt)
#define USE_ASYNC_LDS 1
#endif
#endif
#ifndef USE_ASYNC_LDS
#define USE_ASYNC_LDS 0
#endif

#if USE_ASYNC_LDS
typedef int v4i __attribute__((vector_size(16)));         // matches builtin param element type
typedef __attribute__((address_space(1))) v4i v4i_as1;    // global
typedef __attribute__((address_space(3))) v4i v4i_as3;    // LDS
// two-step cast: reinterpret (pointee/const) in generic AS, then addrspace conversion
#define ASYNC_CP16(gsrc, ldst)                                                  \
    __builtin_amdgcn_global_load_async_to_lds_b128(                             \
        (v4i_as1*)(v4i*)(gsrc), (v4i_as3*)(v4i*)(ldst), 0, 0)
#define WAIT_ASYNC_KEEP3() __builtin_amdgcn_s_wait_asynccnt(3)
#define WAIT_ASYNC_0()     __builtin_amdgcn_s_wait_asynccnt(0)
#else
#define ASYNC_CP16(gsrc, ldst) (*(bf16x8*)(ldst) = *(const bf16x8*)(gsrc))
#define WAIT_ASYNC_KEEP3() ((void)0)
#define WAIT_ASYNC_0()     ((void)0)
#endif

// ---- WMMA helpers (CDNA5 16x16x32 bf16, f32 accum) ----------------------

__device__ __forceinline__ f32x8 wmma_bf16(bf16x16 a, bf16x16 b, f32x8 c) {
    return __builtin_amdgcn_wmma_f32_16x16x32_bf16(false, a, false, b, (short)0, c, false, false);
}

// A fragment: 16x32 bf16 tile, row-major, row stride `ld` elements.
__device__ __forceinline__ bf16x16 frag_a(const __bf16* base, int ld) {
    const int lane = threadIdx.x & 31;
    const int m    = lane & 15;
    const int kh   = (lane >> 4) << 4;        // 0 or 16
    const __bf16* r = base + (size_t)m * ld;
    union { bf16x16 v; bf16x8 h[2]; } u;
    u.h[0] = *(const bf16x8*)(r + kh);
    u.h[1] = *(const bf16x8*)(r + 8 + kh);
    return u.v;
}

// B fragment: 32x16 bf16 tile stored COLUMN-contiguous: colbase[n*ld + k] = B[k][n].
__device__ __forceinline__ bf16x16 frag_b(const __bf16* colbase, int ld) {
    const int lane = threadIdx.x & 31;
    const int n    = lane & 15;
    const int kb   = (lane >> 4) << 4;        // 0 or 16
    return *(const bf16x16*)(colbase + (size_t)n * ld + kb);
}

// C/D 16x16 f32 layout: VGPR r, lanes 0-15 -> (M=r,N=lane); lanes 16-31 -> (M=r+8,N=lane-16).
__device__ __forceinline__ void store_c_f32(float* base, int ld, const f32x8& c) {
    const int lane = threadIdx.x & 31;
    const int n  = lane & 15;
    const int mh = (lane >> 4) << 3;
#pragma unroll
    for (int r = 0; r < 8; ++r) base[(size_t)(mh + r) * ld + n] = c[r];
}

__device__ __forceinline__ void store_c_bf16(__bf16* base, int ld, const f32x8& c) {
    const int lane = threadIdx.x & 31;
    const int n  = lane & 15;
    const int mh = (lane >> 4) << 3;
#pragma unroll
    for (int r = 0; r < 8; ++r) base[(size_t)(mh + r) * ld + n] = (__bf16)c[r];
}

// ---- k0: bf16 casts (src_val, dst_val), W transposes, softplus ----------

__global__ void prep_kernel(const float* __restrict__ src_val,
                            const float* __restrict__ dst_val,
                            const float* __restrict__ src_state,
                            const float* __restrict__ W_route,
                            const float* __restrict__ W_val,
                            __bf16* __restrict__ svb,
                            __bf16* __restrict__ dvb,
                            __bf16* __restrict__ WrT,
                            __bf16* __restrict__ WvT,
                            float*  __restrict__ strength) {
    size_t i = (size_t)blockIdx.x * blockDim.x + threadIdx.x;
    if (i < (size_t)Bn * Sn * DIMn) svb[i] = (__bf16)src_val[i];
    if (i < (size_t)Bn * Dn * DIMn) dvb[i] = (__bf16)dst_val[i];
    if (i < (size_t)DIMn * DIMn) {                 // T[n][k] = W[k][n]
        int n = (int)(i >> 9), k = (int)(i & (DIMn - 1));
        WrT[i] = (__bf16)W_route[(size_t)k * DIMn + n];
        WvT[i] = (__bf16)W_val  [(size_t)k * DIMn + n];
    }
    if (i < (size_t)Bn * Sn) {
        float x = src_state[i];
        strength[i] = fmaxf(x, 0.f) + log1pf(__expf(-fabsf(x)));  // stable softplus
    }
}

// ---- k1: q = src@W_route, pv = src@W_val; emits q row-major + pvT -------
// grid (S/64, DIM/64, B), block 256. Async double-buffered LDS staging.

__global__ void proj_kernel(const __bf16* __restrict__ svb,
                            const __bf16* __restrict__ WrT,
                            const __bf16* __restrict__ WvT,
                            __bf16* __restrict__ qb,
                            __bf16* __restrict__ pvT) {
    const int s0 = blockIdx.x * 64;
    const int n0 = blockIdx.y * 64;
    const int b  = blockIdx.z;
    __shared__ __bf16 As [2][64][40];
    __shared__ __bf16 Bru[2][64][40];
    __shared__ __bf16 Bvu[2][64][40];
    __shared__ __bf16 Cv [64][66];        // pv bounce tile (66 -> conflict-free transpose reads)
    const int t    = threadIdx.x;
    const int wave = t >> 5;
    const int rt   = wave & 3;
    const int cp   = wave >> 2;
    f32x8 accq[2] = {}; f32x8 accv[2] = {};

    // staging coords: 256 threads x one 16B chunk per tile per stage
    const int cr = t >> 2;                 // row 0..63
    const int ck = (t & 3) * 8;            // elem 0,8,16,24
    const __bf16* Ag = svb + ((size_t)b * Sn + s0 + cr) * DIMn + ck;
    const __bf16* Rg = WrT + (size_t)(n0 + cr) * DIMn + ck;
    const __bf16* Vg = WvT + (size_t)(n0 + cr) * DIMn + ck;

    auto stage = [&](int buf, int k0) {
        ASYNC_CP16(Ag + k0, &As [buf][cr][ck]);
        ASYNC_CP16(Rg + k0, &Bru[buf][cr][ck]);
        ASYNC_CP16(Vg + k0, &Bvu[buf][cr][ck]);
    };

    stage(0, 0);
    const int NK = DIMn / 32;
#pragma unroll 1
    for (int i = 0; i < NK; ++i) {
        if (i + 1 < NK) { stage((i + 1) & 1, (i + 1) * 32); WAIT_ASYNC_KEEP3(); }
        else            { WAIT_ASYNC_0(); }
        __syncthreads();
        const int bf = i & 1;
        bf16x16 a = frag_a(&As[bf][rt * 16][0], 40);
#pragma unroll
        for (int c = 0; c < 2; ++c) {
            int ct = cp * 2 + c;
            accq[c] = wmma_bf16(a, frag_b(&Bru[bf][ct * 16][0], 40), accq[c]);
            accv[c] = wmma_bf16(a, frag_b(&Bvu[bf][ct * 16][0], 40), accv[c]);
        }
        __syncthreads();
    }

    // q: row-major bf16 [B,S,DIM]
#pragma unroll
    for (int c = 0; c < 2; ++c) {
        int ct = cp * 2 + c;
        store_c_bf16(qb + ((size_t)b * Sn + s0 + rt * 16) * DIMn + n0 + ct * 16, DIMn, accq[c]);
    }

    // pv: bounce through LDS, emit transposed pvT [B,DIM,S] (coalesced 32B stores)
    {
        const int lane = t & 31;
        const int nn = lane & 15, mh = (lane >> 4) << 3;
#pragma unroll
        for (int c = 0; c < 2; ++c)
#pragma unroll
            for (int r = 0; r < 8; ++r)
                Cv[rt * 16 + mh + r][(cp * 2 + c) * 16 + nn] = (__bf16)accv[c][r];
    }
    __syncthreads();
    {
        const int n = t >> 2, sc = (t & 3) * 16;
        __bf16* dst = pvT + ((size_t)b * DIMn + n0 + n) * Sn + s0 + sc;
        bf16x8 v0, v1;
#pragma unroll
        for (int j = 0; j < 8; ++j) { v0[j] = Cv[sc + j][n]; v1[j] = Cv[sc + 8 + j][n]; }
        *(bf16x8*)(dst) = v0; *(bf16x8*)(dst + 8) = v1;
    }
}

// ---- k2: fused logits GEMM + softmax + transpose + state partials -------
// grid (S/32, B), block 256, dynamic LDS 32xLSTR f32 strip.

__global__ void route_kernel(const __bf16* __restrict__ qb,
                             const __bf16* __restrict__ dvb,
                             const float*  __restrict__ src_state,
                             const float*  __restrict__ strength,
                             __bf16* __restrict__ wT,
                             float*  __restrict__ dspart) {
    extern __shared__ float sm[];
    float* L  = sm;                   // [32][LSTR]
    float* st = sm + 32 * LSTR;
    float* sg = st + 32;
    const int s0   = blockIdx.x * 32;
    const int b    = blockIdx.y;
    const int t    = threadIdx.x;
    const int lane = t & 31;
    const int wave = t >> 5;

    if (t < 32) { st[t] = src_state[(size_t)b * Sn + s0 + t];
                  sg[t] = strength [(size_t)b * Sn + s0 + t]; }

    const int rt = wave & 1;
    const __bf16* Aq = qb + ((size_t)b * Sn + s0 + rt * 16) * DIMn;
#pragma unroll 1
    for (int g = 0; g < 4; ++g) {
        f32x8 acc[4] = {};
        const int ctb = (wave >> 1) * 16 + g * 4;
        for (int k0 = 0; k0 < DIMn; k0 += 32) {
            bf16x16 a = frag_a(Aq + k0, DIMn);
#pragma unroll
            for (int c = 0; c < 4; ++c) {
                const __bf16* Bc = dvb + ((size_t)b * Dn + (ctb + c) * 16) * DIMn + k0;
                acc[c] = wmma_bf16(a, frag_b(Bc, DIMn), acc[c]);
            }
        }
        const int n  = lane & 15;
        const int mh = (lane >> 4) << 3;
#pragma unroll
        for (int c = 0; c < 4; ++c)
#pragma unroll
            for (int r = 0; r < 8; ++r)
                L[(rt * 16 + mh + r) * LSTR + (ctb + c) * 16 + n] = acc[c][r] * INV_SQRT_D;
    }
    __syncthreads();

#pragma unroll 1
    for (int rr = 0; rr < 4; ++rr) {
        const int row = wave * 4 + rr;
        float* Lr = L + row * LSTR;
        float m = -INFINITY;
        for (int d = lane; d < Dn; d += 32) m = fmaxf(m, Lr[d]);
        for (int off = 16; off; off >>= 1) m = fmaxf(m, __shfl_xor(m, off, 32));
        float ssum = 0.f;
        for (int d = lane; d < Dn; d += 32) { float e = __expf(Lr[d] - m); Lr[d] = e; ssum += e; }
        for (int off = 16; off; off >>= 1) ssum += __shfl_xor(ssum, off, 32);
        const float scale = sg[row] / ssum;
        for (int d = lane; d < Dn; d += 32) Lr[d] *= scale;
    }
    __syncthreads();

    for (int d = wave; d < Dn; d += 8)
        wT[((size_t)b * Dn + d) * Sn + s0 + lane] = (__bf16)L[lane * LSTR + d];

    for (int k = t; k < Dn; k += 256) {
        float acc = 0.f;
#pragma unroll
        for (int s = 0; s < 32; ++s) acc += L[s * LSTR + k] * st[s];
        dspart[((size_t)blockIdx.x * Bn + b) * Dn + k] = acc;
    }
}

// ---- k3: delta_val = weightedT @ pv, K = S = 4096 -----------------------
// grid (D/64, DIM/64, B), block 256. Pure global-fragment GEMM: both operands
// are stored K-contiguous (wT rows, pvT rows) -> no LDS, no barriers.

__global__ void outer_kernel(const __bf16* __restrict__ wT,
                             const __bf16* __restrict__ pvT,
                             float* __restrict__ delta_val) {
    const int d0 = blockIdx.x * 64;
    const int n0 = blockIdx.y * 64;
    const int b  = blockIdx.z;
    const int t    = threadIdx.x;
    const int wave = t >> 5;
    const int lane = t & 31;
    const int rt   = wave & 3;
    const int cp   = wave >> 2;
    f32x8 acc[2] = {};
    const __bf16* Ab = wT  + ((size_t)b * Dn   + d0 + rt * 16) * Sn;
    const __bf16* B0 = pvT + ((size_t)b * DIMn + n0 + (cp * 2 + 0) * 16) * Sn;
    const __bf16* B1 = pvT + ((size_t)b * DIMn + n0 + (cp * 2 + 1) * 16) * Sn;

#pragma unroll 1
    for (int k0 = 0; k0 < Sn; k0 += 32) {
        if (k0 + 256 < Sn) {   // stay ahead of L2 latency (global_prefetch_b8)
            const size_t pr = (size_t)(lane & 15) * Sn + k0 + 256;
            __builtin_prefetch(Ab + pr, 0, 1);
            __builtin_prefetch(B0 + pr, 0, 1);
            __builtin_prefetch(B1 + pr, 0, 1);
        }
        bf16x16 a = frag_a(Ab + k0, Sn);
        acc[0] = wmma_bf16(a, frag_b(B0 + k0, Sn), acc[0]);
        acc[1] = wmma_bf16(a, frag_b(B1 + k0, Sn), acc[1]);
    }
#pragma unroll
    for (int c = 0; c < 2; ++c) {
        float* out = delta_val + ((size_t)b * Dn + d0 + rt * 16) * DIMn + n0 + (cp * 2 + c) * 16;
        store_c_f32(out, DIMn, acc[c]);
    }
}

// ---- k4: reduce delta_state partials over the 128 s-blocks --------------

__global__ void ds_reduce_kernel(const float* __restrict__ dspart,
                                 float* __restrict__ delta_state) {
    const int i = blockIdx.x * 256 + threadIdx.x;
    float acc = 0.f;
#pragma unroll 4
    for (int blk = 0; blk < Sn / 32; ++blk) acc += dspart[(size_t)blk * (Bn * Dn) + i];
    delta_state[i] = acc;
}

// ---- host-side launch ----------------------------------------------------

extern "C" void kernel_launch(void* const* d_in, const int* in_sizes, int n_in,
                              void* d_out, int out_size, void* d_ws, size_t ws_size,
                              hipStream_t stream) {
    const float* src_val   = (const float*)d_in[0];
    const float* src_state = (const float*)d_in[1];
    const float* dst_val   = (const float*)d_in[2];
    const float* W_route   = (const float*)d_in[3];
    const float* W_val     = (const float*)d_in[4];

    float* delta_state = (float*)d_out;                       // [B, D]
    float* delta_val   = (float*)d_out + (size_t)Bn * Dn;     // [B, D, DIM]

    char* ws = (char*)d_ws;
    __bf16* qb       = (__bf16*)(ws);                         //  32 MB  [B,S,DIM]
    __bf16* pvT      = (__bf16*)(ws + (32ull  << 20));        //  32 MB  [B,DIM,S]
    __bf16* dvb      = (__bf16*)(ws + (64ull  << 20));        //   8 MB  [B,D,DIM]
    __bf16* wT       = (__bf16*)(ws + (72ull  << 20));        //  64 MB  [B,D,S]
    __bf16* svb      = (__bf16*)(ws + (136ull << 20));        //  32 MB  [B,S,DIM]
    __bf16* WrT      = (__bf16*)(ws + (168ull << 20));        // 512 KB  [DIM,DIM] (transposed)
    __bf16* WvT      = (__bf16*)(ws + (168ull << 20) + (512ull << 10));
    float*  strength = (float*) (ws + (169ull << 20));        // 128 KB  [B,S]
    float*  dspart   = (float*) (ws + (170ull << 20));        //   4 MB  [S/32,B,D]

    prep_kernel<<<(Bn * Sn * DIMn) / 256, 256, 0, stream>>>(
        src_val, dst_val, src_state, W_route, W_val, svb, dvb, WrT, WvT, strength);

    dim3 g1(Sn / 64, DIMn / 64, Bn);
    proj_kernel<<<g1, 256, 0, stream>>>(svb, WrT, WvT, qb, pvT);

    dim3 g2(Sn / 32, Bn);
    size_t shmem2 = (size_t)(32 * LSTR + 64) * sizeof(float);
    route_kernel<<<g2, 256, shmem2, stream>>>(qb, dvb, src_state, strength, wT, dspart);

    dim3 g3(Dn / 64, DIMn / 64, Bn);
    outer_kernel<<<g3, 256, 0, stream>>>(wT, pvT, delta_val);

    ds_reduce_kernel<<<(Bn * Dn) / 256, 256, 0, stream>>>(dspart, delta_state);
}
